// TPA_14267881357669
// MI455X (gfx1250) — matrix operand order repaired and checked by reading, not compiled, not run on hardware
//
#include <hip/hip_runtime.h>
#include <hip/hip_bf16.h>

typedef __bf16 bf16;
typedef __attribute__((ext_vector_type(16))) __bf16 v16bf;
typedef __attribute__((ext_vector_type(8)))  float  v8f;

static __device__ __forceinline__ bf16 f2bf(float f) {
  unsigned u = __builtin_bit_cast(unsigned, f);
  u += 0x7FFFu + ((u >> 16) & 1u);             // round-to-nearest-even
  unsigned short s = (unsigned short)(u >> 16);
  return __builtin_bit_cast(bf16, s);
}

// Build a 16xbf16 fragment from two 16B-aligned 8-element runs -> 2x b128 loads
static __device__ __forceinline__ v16bf load_frag16(const bf16* p0, const bf16* p1) {
  union { uint4 q[2]; v16bf v; } u;
  u.q[0] = *(const uint4*)p0;
  u.q[1] = *(const uint4*)p1;
  return u.v;
}

static __device__ __forceinline__ v16bf load_frag16_u32(const unsigned* p) {
  union { uint4 q[2]; v16bf v; } u;
  u.q[0] = *(const uint4*)p;
  u.q[1] = *(const uint4*)(p + 4);
  return u.v;
}

// CDNA5 async global->LDS copy (16B per lane), tracked by ASYNCcnt.
// LDS address comes from the vdst VGPR; low 32 bits of a generic shared pointer
// are the LDS byte offset.
static __device__ __forceinline__ void async_ld16(const void* g, void* l) {
  asm volatile("global_load_async_to_lds_b128 %0, %1, off"
               :: "v"((unsigned)(size_t)l), "v"((unsigned long long)(size_t)g)
               : "memory");
}
static __device__ __forceinline__ void wait_async0() {
  asm volatile("s_wait_asynccnt 0" ::: "memory");
}

// ---------------------------------------------------------------- converts
__global__ void k_cvt(const float* __restrict__ in, bf16* __restrict__ out, int n) {
  int i = blockIdx.x * blockDim.x + threadIdx.x;
  if (i < n) out[i] = f2bf(in[i]);
}

// Write weight column-major (bf16): dst[(colOff+c)*K + r]. Column-major pairs
// (K even, K odd) in a dword are exactly the WMMA B-fragment dword layout.
__global__ void k_cvt_cat(const float* __restrict__ w, bf16* __restrict__ dst,
                          int K, int Nsrc, int colOff) {
  int i = blockIdx.x * blockDim.x + threadIdx.x;
  if (i < K * Nsrc) {
    int r = i / Nsrc, c = i % Nsrc;
    dst[(size_t)(colOff + c) * K + r] = f2bf(w[i]);
  }
}

// ---------------------------------------------------------------- bf16 GEMM
// C[M,N] = A[M,K](row-major) * W[K,N] (W given column-major packed as dwords).
// Block tile 128x128, K-step 32, double-buffered async LDS staging.
// 8 waves: wave (wm,wn) owns 32Mx64N = 2x4 tiles of 16x16.
#define LDA_S 40   // LDS row stride (elems) for A tile: 80B rows, 16B aligned
#define BSTR  20   // LDS col stride (dwords) for packed B tile: 80B cols

__global__ __launch_bounds__(256) void k_gemm_bf16(
    const bf16* __restrict__ A, const unsigned* __restrict__ Bp,
    float* __restrict__ C, int M, int N, int K) {
  __shared__ __attribute__((aligned(16))) bf16     sA[2][128 * LDA_S];
  __shared__ __attribute__((aligned(16))) unsigned sB[2][128 * BSTR];
  const int tid   = threadIdx.x;
  const int lane  = tid & 31;
  const int wid   = tid >> 5;
  const int wm    = wid & 3;     // 4 row groups of 32
  const int wn    = wid >> 2;    // 2 col groups of 64
  const int bm    = blockIdx.y * 128;
  const int bn    = blockIdx.x * 128;
  const int hh    = lane >> 4;
  const int ln    = lane & 15;
  const int kHalf = K >> 1;      // dwords per column of Bp

  v8f acc[2][4] = {};

  // Per-thread staging assignment (contiguous 32B per side):
  const int srow = tid >> 1;            // A row / B column 0..127
  const int shalf = tid & 1;            // which 16-elem half
  auto stage = [&](int buf, int k0) {
    const bf16* ga = A + (size_t)(bm + srow) * K + k0 + shalf * 16;
    async_ld16(ga,     &sA[buf][srow * LDA_S + shalf * 16]);
    async_ld16(ga + 8, &sA[buf][srow * LDA_S + shalf * 16 + 8]);
    const unsigned* gb = Bp + (size_t)(bn + srow) * kHalf + (k0 >> 1) + shalf * 8;
    async_ld16(gb,     &sB[buf][srow * BSTR + shalf * 8]);
    async_ld16(gb + 4, &sB[buf][srow * BSTR + shalf * 8 + 4]);
  };

  stage(0, 0);
  wait_async0();
  __syncthreads();

  int buf = 0;
  for (int k0 = 0; k0 < K; k0 += 32) {
    if (k0 + 32 < K) stage(buf ^ 1, k0 + 32);

    // A fragments (16x32 bf16, lane half hh holds k = 16*hh + s): two b128 LDS loads each
    v16bf af[2];
#pragma unroll
    for (int mt = 0; mt < 2; ++mt) {
      const bf16* ap = &sA[buf][(wm * 32 + mt * 16 + ln) * LDA_S];
      af[mt] = load_frag16(ap + 16 * hh, ap + 16 * hh + 8);
    }
#pragma unroll
    for (int nt = 0; nt < 4; ++nt) {
      // B fragment: 8 contiguous packed dwords -> two b128 LDS loads
      const int cb = wn * 64 + nt * 16 + ln;
      v16bf bfv = load_frag16_u32(&sB[buf][cb * BSTR + 8 * hh]);
#pragma unroll
      for (int mt = 0; mt < 2; ++mt)
        acc[mt][nt] = __builtin_amdgcn_wmma_f32_16x16x32_bf16(
            false, af[mt], false, bfv, (short)0, acc[mt][nt], false, false);
    }
    wait_async0();
    __syncthreads();
    buf ^= 1;
  }
#pragma unroll
  for (int mt = 0; mt < 2; ++mt)
#pragma unroll
    for (int nt = 0; nt < 4; ++nt) {
      int col = bn + wn * 64 + nt * 16 + ln;
#pragma unroll
      for (int j = 0; j < 8; ++j) {
        int row = bm + wm * 32 + mt * 16 + j + 8 * hh;
        C[(size_t)row * N + col] = acc[mt][nt][j];
      }
    }
}

// ---------------------------------------------------- RoPE + rank contraction
// Y[t, 0:192]=A_q(H,QR) 192:224=A_k 224:256=A_v 256:1792=B_q(12,128)
// 1792:2048=B_k(2,128) 2048:2304=B_v(2,128)
__global__ __launch_bounds__(64) void k_rope_contract(
    const float* __restrict__ Y, const float* __restrict__ cosp,
    const float* __restrict__ sinp,
    bf16* __restrict__ qh,   // [B*H][S][128] (scaled by 1/QR/sqrt(HD))
    bf16* __restrict__ kh,   // [B*H][S][128] (scaled by 1/R)
    bf16* __restrict__ vt) { // [B*H][128][S] transposed (scaled by 1/R)
  __shared__ float sc[256];
  const int t = blockIdx.x;
  const int b = t >> 11;
  const int s = t & 2047;
  const int p = threadIdx.x;   // rotary pair 0..63
  const float* y = Y + (size_t)t * 2304;
  for (int i = p; i < 256; i += 64) sc[i] = y[i];
  __syncthreads();
  const float cv = cosp[s * 64 + p];
  const float sv = sinp[s * 64 + p];

  float bq0[12], bq1[12];
#pragma unroll
  for (int r = 0; r < 12; ++r) {
    float t0 = y[256 + r * 128 + 2 * p], t1 = y[256 + r * 128 + 2 * p + 1];
    bq0[r] = t0 * cv - t1 * sv;
    bq1[r] = t0 * sv + t1 * cv;
  }
  float bk0[2], bk1[2], bv0[2], bv1[2];
#pragma unroll
  for (int r = 0; r < 2; ++r) {
    float t0 = y[1792 + r * 128 + 2 * p], t1 = y[1792 + r * 128 + 2 * p + 1];
    bk0[r] = t0 * cv - t1 * sv;
    bk1[r] = t0 * sv + t1 * cv;
    bv0[r] = y[2048 + r * 128 + 2 * p];
    bv1[r] = y[2048 + r * 128 + 2 * p + 1];
  }
  const float qsc  = (1.0f / 12.0f) * 0.08838834764831845f; // 1/QR * 1/sqrt(128)
  const float kvsc = 0.5f;                                   // 1/R
  for (int h = 0; h < 16; ++h) {
    float q0 = 0.f, q1 = 0.f;
#pragma unroll
    for (int r = 0; r < 12; ++r) {
      float a = sc[h * 12 + r];
      q0 += a * bq0[r]; q1 += a * bq1[r];
    }
    float ak0 = sc[192 + h * 2], ak1 = sc[192 + h * 2 + 1];
    float av0 = sc[224 + h * 2], av1 = sc[224 + h * 2 + 1];
    float k0v = (ak0 * bk0[0] + ak1 * bk0[1]) * kvsc;
    float k1v = (ak0 * bk1[0] + ak1 * bk1[1]) * kvsc;
    float v0v = (av0 * bv0[0] + av1 * bv0[1]) * kvsc;
    float v1v = (av0 * bv1[0] + av1 * bv1[1]) * kvsc;
    size_t bh = (size_t)(b * 16 + h);
    size_t qi = (bh * 2048 + s) * 128 + 2 * p;
    qh[qi] = f2bf(q0 * qsc); qh[qi + 1] = f2bf(q1 * qsc);
    kh[qi] = f2bf(k0v);      kh[qi + 1] = f2bf(k1v);
    vt[(bh * 128 + 2 * p) * 2048 + s]     = f2bf(v0v);
    vt[(bh * 128 + 2 * p + 1) * 2048 + s] = f2bf(v1v);
  }
}

// ---------------------------------------------------------- flash attention
// One wave per (b,h, 16-query tile). Streaming softmax over 32-key blocks.
#define SP_S 40   // P-tile LDS row stride (elems): 80B rows -> aligned b128

__global__ __launch_bounds__(32) void k_flash(
    const bf16* __restrict__ qh, const bf16* __restrict__ kh,
    const bf16* __restrict__ vt, bf16* __restrict__ attn) {
  __shared__ __attribute__((aligned(16))) bf16 sp[16 * SP_S]; // P tile 16x32
  const int qt   = blockIdx.x;     // 0..127
  const int bh   = blockIdx.y;     // 0..31
  const int b    = bh >> 4, h = bh & 15;
  const int lane = threadIdx.x;
  const int hh   = lane >> 4, ln = lane & 15;
  const bf16* qp = qh + (size_t)bh * 2048 * 128;
  const bf16* kp = kh + (size_t)bh * 2048 * 128;
  const bf16* vp = vt + (size_t)bh * 128 * 2048;

  v16bf qf[4];
  {
    const bf16* qrow = qp + (size_t)(qt * 16 + ln) * 128;
#pragma unroll
    for (int c = 0; c < 4; ++c)
      qf[c] = load_frag16(qrow + c * 32 + 16 * hh, qrow + c * 32 + 16 * hh + 8);
  }
  v8f o[8] = {};
  float mr[8], lr[8];
#pragma unroll
  for (int j = 0; j < 8; ++j) { mr[j] = -3.0e38f; lr[j] = 0.f; }

  const int kEnd = qt * 16 + 16;
  for (int kb = 0; kb < kEnd; kb += 32) {
    v8f s0 = {}, s1 = {};
#pragma unroll
    for (int c = 0; c < 4; ++c) {
      const bf16* p0 = kp + (size_t)(kb + ln) * 128 + c * 32 + 16 * hh;
      const bf16* p1 = kp + (size_t)(kb + 16 + ln) * 128 + c * 32 + 16 * hh;
      v16bf kf0 = load_frag16(p0, p0 + 8);
      v16bf kf1 = load_frag16(p1, p1 + 8);
      s0 = __builtin_amdgcn_wmma_f32_16x16x32_bf16(false, qf[c], false, kf0, (short)0, s0, false, false);
      s1 = __builtin_amdgcn_wmma_f32_16x16x32_bf16(false, qf[c], false, kf1, (short)0, s1, false, false);
    }
    // causal mask + row max (16-lane half reductions)
    float pm[8];
#pragma unroll
    for (int j = 0; j < 8; ++j) {
      int qrow = qt * 16 + j + 8 * hh;
      float a = s0[j], bb = s1[j];
      if (kb + ln > qrow)      a  = -1.0e9f;
      if (kb + 16 + ln > qrow) bb = -1.0e9f;
      s0[j] = a; s1[j] = bb;
      float v = fmaxf(a, bb);
      v = fmaxf(v, __shfl_xor(v, 1, 32));
      v = fmaxf(v, __shfl_xor(v, 2, 32));
      v = fmaxf(v, __shfl_xor(v, 4, 32));
      v = fmaxf(v, __shfl_xor(v, 8, 32));
      pm[j] = v;
    }
    float fsc[8];
#pragma unroll
    for (int j = 0; j < 8; ++j) {
      float mn = fmaxf(mr[j], pm[j]);
      fsc[j] = __expf(mr[j] - mn);
      float p0 = __expf(s0[j] - mn);
      float p1 = __expf(s1[j] - mn);
      s0[j] = p0; s1[j] = p1;
      float rs = p0 + p1;
      rs += __shfl_xor(rs, 1, 32);
      rs += __shfl_xor(rs, 2, 32);
      rs += __shfl_xor(rs, 4, 32);
      rs += __shfl_xor(rs, 8, 32);
      lr[j] = lr[j] * fsc[j] + rs;
      mr[j] = mn;
    }
#pragma unroll
    for (int dc = 0; dc < 8; ++dc)
#pragma unroll
      for (int j = 0; j < 8; ++j) o[dc][j] *= fsc[j];

    // P: D-layout -> A-layout via LDS
#pragma unroll
    for (int j = 0; j < 8; ++j) {
      sp[(j + 8 * hh) * SP_S + ln]      = f2bf(s0[j]);
      sp[(j + 8 * hh) * SP_S + 16 + ln] = f2bf(s1[j]);
    }
    __syncthreads();
    v16bf pf = load_frag16(&sp[ln * SP_S + 16 * hh], &sp[ln * SP_S + 16 * hh + 8]);
    __syncthreads();
    // P @ V (vT makes B-fragments contiguous)
#pragma unroll
    for (int dc = 0; dc < 8; ++dc) {
      const bf16* pr = vp + (size_t)(dc * 16 + ln) * 2048 + kb + 16 * hh;
      v16bf vf = load_frag16(pr, pr + 8);
      o[dc] = __builtin_amdgcn_wmma_f32_16x16x32_bf16(false, pf, false, vf, (short)0, o[dc], false, false);
    }
  }
  float inv[8];
#pragma unroll
  for (int j = 0; j < 8; ++j) inv[j] = 1.0f / lr[j];
#pragma unroll
  for (int dc = 0; dc < 8; ++dc)
#pragma unroll
    for (int j = 0; j < 8; ++j) {
      int srow = qt * 16 + j + 8 * hh;
      size_t oi = ((size_t)(b * 2048 + srow)) * 2048 + h * 128 + dc * 16 + ln;
      attn[oi] = f2bf(o[dc][j] * inv[j]);
    }
}

// ------------------------------------------------------------------- launch
extern "C" void kernel_launch(void* const* d_in, const int* in_sizes, int n_in,
                              void* d_out, int out_size, void* d_ws, size_t ws_size,
                              hipStream_t stream) {
  const float* x   = (const float*)d_in[0];
  const float* fc  = (const float*)d_in[1];
  const float* fs  = (const float*)d_in[2];
  // d_in[3] mask unused: causal mask applied analytically
  const float* wAq = (const float*)d_in[4];
  const float* wAk = (const float*)d_in[5];
  const float* wAv = (const float*)d_in[6];
  const float* wBq = (const float*)d_in[7];
  const float* wBk = (const float*)d_in[8];
  const float* wBv = (const float*)d_in[9];
  const float* wO  = (const float*)d_in[10];
  (void)in_sizes; (void)n_in; (void)out_size; (void)ws_size;

  char* ws = (char*)d_ws;
  size_t off = 0;
  bf16*  x_bf  = (bf16*)(ws + off); off += (size_t)4096 * 2048 * 2;
  bf16*  wcat  = (bf16*)(ws + off); off += (size_t)2048 * 2304 * 2; // col-major
  bf16*  wo_bf = (bf16*)(ws + off); off += (size_t)2048 * 2048 * 2; // col-major
  float* Y     = (float*)(ws + off); off += (size_t)4096 * 2304 * 4;
  bf16*  q_bf  = (bf16*)(ws + off); off += (size_t)32 * 2048 * 128 * 2;
  bf16*  k_bf  = (bf16*)(ws + off); off += (size_t)32 * 2048 * 128 * 2;
  bf16*  vT_bf = (bf16*)(ws + off); off += (size_t)32 * 128 * 2048 * 2;
  bf16*  at_bf = (bf16*)(ws + off); off += (size_t)4096 * 2048 * 2;

  int n = 4096 * 2048;
  k_cvt<<<(n + 255) / 256, 256, 0, stream>>>(x, x_bf, n);
  k_cvt_cat<<<(2048 * 192 + 255) / 256, 256, 0, stream>>>(wAq, wcat, 2048, 192, 0);
  k_cvt_cat<<<(2048 * 32 + 255) / 256, 256, 0, stream>>>(wAk, wcat, 2048, 32, 192);
  k_cvt_cat<<<(2048 * 32 + 255) / 256, 256, 0, stream>>>(wAv, wcat, 2048, 32, 224);
  k_cvt_cat<<<(2048 * 1536 + 255) / 256, 256, 0, stream>>>(wBq, wcat, 2048, 1536, 256);
  k_cvt_cat<<<(2048 * 256 + 255) / 256, 256, 0, stream>>>(wBk, wcat, 2048, 256, 1792);
  k_cvt_cat<<<(2048 * 256 + 255) / 256, 256, 0, stream>>>(wBv, wcat, 2048, 256, 2048);
  k_cvt_cat<<<(2048 * 2048 + 255) / 256, 256, 0, stream>>>(wO, wo_bf, 2048, 2048, 0);

  // Fused projection GEMM: [4096,2048] x [2048,2304] -> Y fp32
  k_gemm_bf16<<<dim3(2304 / 128, 4096 / 128), 256, 0, stream>>>(
      x_bf, (const unsigned*)wcat, Y, 4096, 2304, 2048);
  // RoPE + rank contraction -> q,k,[vT] bf16
  k_rope_contract<<<4096, 64, 0, stream>>>(Y, fc, fs, q_bf, k_bf, vT_bf);
  // Flash attention -> attn bf16 [b,s,h*128]
  k_flash<<<dim3(128, 32), 32, 0, stream>>>(q_bf, k_bf, vT_bf, at_bf);
  // Output projection: [4096,2048] x [2048,2048] -> d_out fp32
  k_gemm_bf16<<<dim3(2048 / 128, 4096 / 128), 256, 0, stream>>>(
      at_bf, (const unsigned*)wo_bf, (float*)d_out, 4096, 2048, 2048);
}
